// ArcFaceLoss_82085414961612
// MI455X (gfx1250) — compile-verified, matrix operand order checked by reading, not run on hardware
//
#include <hip/hip_runtime.h>
#include <math.h>

// ---------------- problem constants ----------------
#define BATCH      1024
#define EMB        512
#define NCLS       100000
#define BM         128
#define BN         128
#define BK         32
#define NBM        (BATCH / BM)              // 8
#define NBN        ((NCLS + BN - 1) / BN)    // 782
#define LDT        40                        // padded LDS row stride (elements)
#define BIG_NEG    (-3.0e38f)

#define ARC_COSM   0.8775825618903728f
#define ARC_SINM   0.479425538604203f
#define ARC_TH     (-0.8775825618903728f)
#define ARC_MM     0.2397127693021015f
#define ARC_SCALE  64.0f

#define AS1 __attribute__((address_space(1)))
#define AS3 __attribute__((address_space(3)))

typedef __attribute__((ext_vector_type(16))) __bf16 v16bf;
typedef __attribute__((ext_vector_type(8)))  float  v8f;
typedef int v4i_t __attribute__((vector_size(16)));   // matches builtin param

typedef unsigned short ushort_t;
typedef unsigned int   uint_t;

#if __has_builtin(__builtin_amdgcn_global_load_async_to_lds_b128)
#define USE_ASYNC_LDS 1
#else
#define USE_ASYNC_LDS 0
#endif

// f32 -> bf16, round-to-nearest-even
static __device__ __forceinline__ ushort_t f2bf(float f) {
    union { float f; uint_t u; } v; v.f = f;
    uint_t r = v.u + 0x7FFFu + ((v.u >> 16) & 1u);
    return (ushort_t)(r >> 16);
}
static __device__ __forceinline__ float bf2f(ushort_t h) {
    union { uint_t u; float f; } v; v.u = ((uint_t)h) << 16;
    return v.f;
}

// raw v_sqrt_f32 (approx is fine: feeding sqrt(clip(1-c^2)) into the margin)
static __device__ __forceinline__ float fast_sqrtf(float x) {
#if __has_builtin(__builtin_amdgcn_sqrtf)
    return __builtin_amdgcn_sqrtf(x);
#else
    return sqrtf(x);
#endif
}

template <int N>
static __device__ __forceinline__ void wait_asynccnt() {
#if __has_builtin(__builtin_amdgcn_s_wait_asynccnt)
    __builtin_amdgcn_s_wait_asynccnt((unsigned short)N);
#else
    asm volatile("s_wait_asynccnt %0" ::"i"(N));
#endif
}

// ---------------------------------------------------------------------------
// Kernel 1: L2-normalize a row of 512 floats, emit bf16.  grid = nrows, blk=128
// ---------------------------------------------------------------------------
__global__ void rownorm_to_bf16(const float* __restrict__ src,
                                ushort_t* __restrict__ dst) {
    const int r = blockIdx.x;
    const int t = threadIdx.x;                 // 0..127, 4 elems each
    const float4 x = ((const float4*)(src + (size_t)r * EMB))[t];
    float ss = x.x * x.x + x.y * x.y + x.z * x.z + x.w * x.w;

    __shared__ float red[128];
    red[t] = ss;
    __syncthreads();
#pragma unroll
    for (int off = 64; off > 0; off >>= 1) {
        if (t < off) red[t] += red[t + off];
        __syncthreads();
    }
    const float inv = 1.0f / fmaxf(fast_sqrtf(red[0]), 1e-12f);

    uint_t lo = (uint_t)f2bf(x.x * inv) | ((uint_t)f2bf(x.y * inv) << 16);
    uint_t hi = (uint_t)f2bf(x.z * inv) | ((uint_t)f2bf(x.w * inv) << 16);
    uint2 packed; packed.x = lo; packed.y = hi;
    ((uint2*)(dst + (size_t)r * EMB))[t] = packed;
}

// ---------------------------------------------------------------------------
// Fragment loader: 16-bit A/B WMMA operand from LDS.
// lanes 0-15 : row = base+lane,    K 0-7 (q0) and 16-23 (q1)
// lanes 16-31: row = base+lane-16, K 8-15 (q0) and 24-31 (q1)
// ---------------------------------------------------------------------------
static __device__ __forceinline__ v16bf load_frag(const ushort_t* sm,
                                                  int rowbase, int lane) {
    const int r    = rowbase + (lane & 15);
    const int half = (lane >> 4) << 3;         // 0 or 8
    const ushort_t* p = sm + r * LDT + half;
    union { uint4 q[2]; v16bf v; } u;
    u.q[0] = *(const uint4*)(p);
    u.q[1] = *(const uint4*)(p + 16);
    return u.v;
}

// streaming (m,s) combine helper
static __device__ __forceinline__ void lse_combine(float& m, float& s,
                                                   float mo, float so) {
    const float M = fmaxf(m, mo);
    s = s * __expf(m - M) + so * __expf(mo - M);
    m = M;
}

// ---------------------------------------------------------------------------
// Kernel 2: bf16 WMMA cosine GEMM + ArcFace margin + per-block log-sum-exp.
// grid = (NBN, NBM), block = 256 threads (8 wave32: 4 M-waves x 2 N-waves)
// Tiles are double-buffered in LDS and filled by the gfx1250 async-to-LDS
// engine (ASYNCcnt), overlapping the next tile's fill with WMMA compute.
// ---------------------------------------------------------------------------
__global__ __launch_bounds__(256)
void arcface_gemm_lse(const ushort_t* __restrict__ ebf,
                      const ushort_t* __restrict__ wbf,
                      const int* __restrict__ labels,
                      float* __restrict__ part_m,
                      float* __restrict__ part_s) {
    __shared__ __align__(16) ushort_t smA[2][BM * LDT];
    __shared__ __align__(16) ushort_t smB[2][BN * LDT];
    __shared__ int   labS[BM];
    __shared__ float redm[2][BM];
    __shared__ float reds[2][BM];

    const int tid    = threadIdx.x;
    const int lane   = tid & 31;
    const int wid    = tid >> 5;
    const int wm     = wid >> 1;     // 0..3  -> 32 rows each
    const int wn     = wid & 1;      // 0..1  -> 64 cols each
    const int m_base = blockIdx.y * BM;
    const int n_base = blockIdx.x * BN;

    if (tid < BM) labS[tid] = labels[m_base + tid];

    // per-thread tile-chunk mapping: 512 x 16B chunks, 2 per thread
    const ushort_t* gA[2];
    const ushort_t* gB[2];
    int ldsOff[2];
#pragma unroll
    for (int i = 0; i < 2; ++i) {
        const int chunk = tid + i * 256;         // 0..511
        const int row   = chunk >> 2;            // 0..127
        const int col   = (chunk & 3) * 8;       // 0,8,16,24
        gA[i] = ebf + (size_t)(m_base + row) * EMB + col;
        int cls = n_base + row;
        if (cls >= NCLS) cls = NCLS - 1;         // clamp: garbage masked later
        gB[i] = wbf + (size_t)cls * EMB + col;
        ldsOff[i] = row * LDT + col;
    }

    v8f acc[2][4] = {};

#if USE_ASYNC_LDS
    auto issue_tile = [&](int buf, int kt) {
#pragma unroll
        for (int i = 0; i < 2; ++i) {
            __builtin_amdgcn_global_load_async_to_lds_b128(
                (AS1 v4i_t*)(gA[i] + kt),
                (AS3 v4i_t*)&smA[buf][ldsOff[i]], 0, 0);
            __builtin_amdgcn_global_load_async_to_lds_b128(
                (AS1 v4i_t*)(gB[i] + kt),
                (AS3 v4i_t*)&smB[buf][ldsOff[i]], 0, 0);
        }
    };
    issue_tile(0, 0);
#endif

    int cur = 0;
    for (int kt = 0; kt < EMB; kt += BK, cur ^= 1) {
#if USE_ASYNC_LDS
        __syncthreads();                 // buf[cur^1] no longer being read
        if (kt + BK < EMB) {
            issue_tile(cur ^ 1, kt + BK);
            wait_asynccnt<4>();          // own 4 loads of stage `cur` retired
        } else {
            wait_asynccnt<0>();
        }
        __syncthreads();                 // every wave's stage tiles visible
#else
        uint4 areg[2], breg[2];
#pragma unroll
        for (int i = 0; i < 2; ++i) {
            areg[i] = *(const uint4*)(gA[i] + kt);
            breg[i] = *(const uint4*)(gB[i] + kt);
        }
        __syncthreads();                 // prior compute done
#pragma unroll
        for (int i = 0; i < 2; ++i) {
            *(uint4*)(&smA[cur][ldsOff[i]]) = areg[i];
            *(uint4*)(&smB[cur][ldsOff[i]]) = breg[i];
        }
        __syncthreads();                 // tiles visible
#endif
        v16bf af[2];
#pragma unroll
        for (int sm = 0; sm < 2; ++sm)
            af[sm] = load_frag(smA[cur], wm * 32 + sm * 16, lane);
#pragma unroll
        for (int sn = 0; sn < 4; ++sn) {
            const v16bf bf_ = load_frag(smB[cur], wn * 64 + sn * 16, lane);
#pragma unroll
            for (int sm = 0; sm < 2; ++sm) {
                acc[sm][sn] = __builtin_amdgcn_wmma_f32_16x16x32_bf16(
                    false, af[sm], false, bf_, (short)0, acc[sm][sn],
                    false, false);
            }
        }
    }

    // ------- epilogue: margin + scale + per-(row, block) log-sum-exp -------
#pragma unroll
    for (int sm = 0; sm < 2; ++sm) {
#pragma unroll
        for (int v = 0; v < 8; ++v) {
            const int rowL = wm * 32 + sm * 16 + v + ((lane >> 4) << 3);
            const int lab  = labS[rowL];
            float l[4];
#pragma unroll
            for (int sn = 0; sn < 4; ++sn) {
                const int col = n_base + wn * 64 + sn * 16 + (lane & 15);
                const float c = acc[sm][sn][v];
                float s2 = 1.0f - c * c;
                s2 = fminf(fmaxf(s2, 0.0f), 1.0f);
                const float sine = fast_sqrtf(s2);
                float phi = c * ARC_COSM - sine * ARC_SINM;
                phi = (c > ARC_TH) ? phi : (c - ARC_MM);
                const float val = (col == lab) ? phi : c;
                l[sn] = (col < NCLS) ? (ARC_SCALE * val) : BIG_NEG;
            }
            float m = fmaxf(fmaxf(l[0], l[1]), fmaxf(l[2], l[3]));
            float s = __expf(l[0] - m) + __expf(l[1] - m) +
                      __expf(l[2] - m) + __expf(l[3] - m);
            // reduce across the 16-lane half-group (same row, 16 columns)
#pragma unroll
            for (int d = 1; d < 16; d <<= 1) {
                const float mo = __shfl_xor(m, d, 32);
                const float so = __shfl_xor(s, d, 32);
                lse_combine(m, s, mo, so);
            }
            if ((lane & 15) == 0) {
                redm[wn][rowL] = m;
                reds[wn][rowL] = s;
            }
        }
    }
    __syncthreads();
    if (tid < BM) {
        float m = redm[0][tid], s = reds[0][tid];
        lse_combine(m, s, redm[1][tid], reds[1][tid]);
        const size_t idx = (size_t)(m_base + tid) * NBN + blockIdx.x;
        part_m[idx] = m;
        part_s[idx] = s;
    }
}

// ---------------------------------------------------------------------------
// Kernel 3: per-row final reduce: label logit + merge NBN partials -> mean NLL
// grid = BATCH, block = 256
// ---------------------------------------------------------------------------
__global__ __launch_bounds__(256)
void arcface_finalize(const ushort_t* __restrict__ ebf,
                      const ushort_t* __restrict__ wbf,
                      const int* __restrict__ labels,
                      const float* __restrict__ part_m,
                      const float* __restrict__ part_s,
                      float* __restrict__ out) {
    const int b = blockIdx.x;
    const int t = threadIdx.x;
    __shared__ float ra[256];
    __shared__ float rb[256];

    // label cosine from the same bf16 data (consistent with GEMM values)
    const int lab = labels[b];
    float d = 0.0f;
#pragma unroll
    for (int k = t; k < EMB; k += 256)
        d += bf2f(ebf[(size_t)b * EMB + k]) * bf2f(wbf[(size_t)lab * EMB + k]);
    ra[t] = d;
    __syncthreads();
#pragma unroll
    for (int off = 128; off > 0; off >>= 1) {
        if (t < off) ra[t] += ra[t + off];
        __syncthreads();
    }
    const float cosl = ra[0];
    __syncthreads();

    // merge the NBN streaming-softmax partials of this row
    float m = BIG_NEG, s = 0.0f;
    for (int i = t; i < NBN; i += 256) {
        const float pm = part_m[(size_t)b * NBN + i];
        const float ps = part_s[(size_t)b * NBN + i];
        lse_combine(m, s, pm, ps);
    }
    ra[t] = m; rb[t] = s;
    __syncthreads();
#pragma unroll
    for (int off = 128; off > 0; off >>= 1) {
        if (t < off) {
            float mm = ra[t], ss = rb[t];
            lse_combine(mm, ss, ra[t + off], rb[t + off]);
            ra[t] = mm; rb[t] = ss;
        }
        __syncthreads();
    }

    if (t == 0) {
        const float logZ = ra[0] + logf(rb[0]);
        float s2 = 1.0f - cosl * cosl;
        s2 = fminf(fmaxf(s2, 0.0f), 1.0f);
        const float sine = fast_sqrtf(s2);
        float phi = cosl * ARC_COSM - sine * ARC_SINM;
        phi = (cosl > ARC_TH) ? phi : (cosl - ARC_MM);
        const float nll = logZ - ARC_SCALE * phi;
        atomicAdd(out, nll * (1.0f / (float)BATCH));
    }
}

// ---------------------------------------------------------------------------
// Launch
// ---------------------------------------------------------------------------
extern "C" void kernel_launch(void* const* d_in, const int* in_sizes, int n_in,
                              void* d_out, int out_size, void* d_ws, size_t ws_size,
                              hipStream_t stream) {
    (void)in_sizes; (void)n_in; (void)out_size; (void)ws_size;
    const float* emb    = (const float*)d_in[0];
    const float* weight = (const float*)d_in[1];
    const int*   labels = (const int*)d_in[2];
    float*       out    = (float*)d_out;

    // workspace layout
    char* ws = (char*)d_ws;
    constexpr size_t EBF_BYTES = (size_t)BATCH * EMB * 2;          //   1.0 MB
    constexpr size_t WBF_BYTES = (size_t)NCLS  * EMB * 2;          // 102.4 MB
    constexpr size_t PM_BYTES  = (size_t)BATCH * NBN * 4;          //   3.2 MB
    ushort_t* ebf = (ushort_t*)ws;
    ushort_t* wbf = (ushort_t*)(ws + EBF_BYTES);
    float*    pm  = (float*)(ws + EBF_BYTES + WBF_BYTES);
    float*    psum= (float*)(ws + EBF_BYTES + WBF_BYTES + PM_BYTES);

    // 1) normalize + cast to bf16 (W becomes 102 MB -> fully L2-resident)
    rownorm_to_bf16<<<BATCH, 128, 0, stream>>>(emb, ebf);
    rownorm_to_bf16<<<NCLS,  128, 0, stream>>>(weight, wbf);

    // 2) WMMA cosine GEMM + ArcFace margin + blockwise log-sum-exp partials
    dim3 grid(NBN, NBM);
    arcface_gemm_lse<<<grid, 256, 0, stream>>>(ebf, wbf, labels, pm, psum);

    // 3) final per-row reduce -> mean loss scalar
    (void)hipMemsetAsync(d_out, 0, sizeof(float), stream);
    arcface_finalize<<<BATCH, 256, 0, stream>>>(ebf, wbf, labels, pm, psum, out);
}
